// GeneticAlgorithm_22763326669404
// MI455X (gfx1250) — matrix operand order repaired and checked by reading, not compiled, not run on hardware
//
#include <hip/hip_runtime.h>
#include <math.h>

typedef __attribute__((ext_vector_type(16))) _Float16 v16h;
typedef __attribute__((ext_vector_type(8)))  _Float16 v8h;
typedef __attribute__((ext_vector_type(2)))  _Float16 v2h;
typedef __attribute__((ext_vector_type(8)))  float    v8f;

#define P_POP   210
#define D_DIM   768
#define KATTN   40
#define KMUT    20
#define MPAD    48          // 40 selected rows padded to 3 tiles of 16
#define NPAD    256         // 210 cols padded to 16 tiles of 16 (48-tile grid = 6/wave)
#define KSLAB   64          // K-slab staged in LDS as f16
#define MAT_ELEMS (P_POP * D_DIM)

// ---------------- deterministic RNG helpers ----------------
__device__ __forceinline__ unsigned long long splitmix64(unsigned long long x) {
    x += 0x9E3779B97F4A7C15ULL;
    x = (x ^ (x >> 30)) * 0xBF58476D1CE4E5B9ULL;
    x = (x ^ (x >> 27)) * 0x94D049BB133111EBULL;
    return x ^ (x >> 31);
}
__device__ __forceinline__ float u01(unsigned long long h) {
    return ((float)((unsigned)(h >> 40)) + 0.5f) * (1.0f / 16777216.0f);
}

// ---------------- attention stage kernel ----------------
// One workgroup (256 threads = 8 waves) per attention call.
// stage 0: c1 = attn(p1, p2) from inputs. stage 1: c2 = attn(p2, c1), c1 in d_out.
__global__ __launch_bounds__(256) void attn_kernel(const float* __restrict__ in,
                                                   float* __restrict__ out,
                                                   int stage) {
    // LDS plan: phase 1 uses f16 staging panels Af[48][64] + Bf[256][64] (39 KB);
    // phase 2 reuses the same bytes as the f32 score panel S[48][256] (48 KB).
    __shared__ __align__(16) unsigned char lsbuf[MPAD * NPAD * 4];
    __shared__ int   ridx[KATTN];
    __shared__ short pool[P_POP];

    _Float16* Af = (_Float16*)lsbuf;                            // [MPAD][KSLAB]
    _Float16* Bf = (_Float16*)(lsbuf + MPAD * KSLAB * 2);       // [NPAD][KSLAB]
    float*    S  = (float*)lsbuf;                               // [MPAD][NPAD]

    const int id  = blockIdx.x;        // 0..63
    const int blk = id >> 2;           // block 0..15
    const int cr  = id & 3;            // crossover 0..3 within block

    const int a0 = 4 * blk, a1 = a0 + 1, a2 = a0 + 2, a3 = a0 + 3;
    const int b0 = 64 + 4 * blk, b1 = b0 + 1, b2 = b0 + 2, b3 = b0 + 3;

    int iA, iB, iO, key, bFromOut;
    if (stage == 0) {
        bFromOut = 0;
        switch (cr) {
            case 0:  iA = a0; iB = b0; iO = a0; key = 0; break;
            case 1:  iA = b1; iB = a1; iO = b1; key = 2; break;
            case 2:  iA = a2; iB = a3; iO = a2; key = 4; break;
            default: iA = b2; iB = b3; iO = b2; key = 6; break;
        }
    } else {
        bFromOut = 1;                  // B operand is the stage-0 child, already in d_out
        switch (cr) {
            case 0:  iA = b0; iB = a0; iO = b0; key = 1; break;
            case 1:  iA = a1; iB = b1; iO = a1; key = 3; break;
            case 2:  iA = a3; iB = a2; iO = a3; key = 5; break;
            default: iA = b3; iB = b2; iO = b3; key = 7; break;
        }
    }
    const float* A = in + (size_t)iA * MAT_ELEMS;
    const float* B = (bFromOut ? (const float*)out : in) + (size_t)iB * MAT_ELEMS;
    float*       O = out + (size_t)iO * MAT_ELEMS;

    const int tid  = threadIdx.x;
    const int wave = tid >> 5;
    const int lane = tid & 31;

    // ---- pick 40 distinct rows (partial Fisher-Yates), deterministic per (blk,key) ----
    if (tid == 0) {
        for (int i = 0; i < P_POP; ++i) pool[i] = (short)i;
        unsigned long long s =
            splitmix64(0xA5D1C0DEULL ^ ((unsigned long long)blk << 16) ^ (unsigned long long)key);
        for (int t = 0; t < KATTN; ++t) {
            s = splitmix64(s);
            int j = t + (int)(s % (unsigned long long)(P_POP - t));
            short tmp = pool[t]; pool[t] = pool[j]; pool[j] = tmp;
            ridx[t] = pool[t];
        }
    }

    // ---- copy A -> O (fusion overwrites the 40 selected rows later) ----
    {
        const float4* s4 = (const float4*)A;
        float4*       d4 = (float4*)O;
        for (int i = tid; i < MAT_ELEMS / 4; i += 256) d4[i] = s4[i];
    }
    __syncthreads();   // ridx visible; staging may begin

    // ---- register-blocked ownership: wave w owns tm in {0,1,2} x tn in {w, w+8} ----
    v8f acc[3][2];
#pragma unroll
    for (int tm = 0; tm < 3; ++tm)
#pragma unroll
        for (int j = 0; j < 2; ++j) acc[tm][j] = (v8f){};

    const int mloc = lane & 15;
    const int khiA = (lane >> 4) * 16;   // A frag: lanes 16..31 carry K+16
    const int khiB = (lane >> 4) * 16;   // B frag: lanes 16..31 carry K+16

    // ---- K-slab staged WMMA GEMM: S[m,n] = A[ridx[m],:] . B[n,:] ----
    for (int kb = 0; kb < D_DIM; kb += KSLAB) {
        // stage + convert A_sel slab (48 x 64) once per element
        for (int e = tid; e < MPAD * (KSLAB / 2); e += 256) {
            const int row = e >> 5, pr = (e & 31) * 2;
            const int arow = (row < KATTN) ? ridx[row] : 0;      // pad rows: harmless dup
            const float2 f = *(const float2*)(A + (size_t)arow * D_DIM + kb + pr);
            v2h h; h.x = (_Float16)f.x; h.y = (_Float16)f.y;
            *(v2h*)(Af + row * KSLAB + pr) = h;
        }
        // stage + convert B slab (256 x 64); rows >= 210 clamped (masked in softmax)
        for (int e = tid; e < NPAD * (KSLAB / 2); e += 256) {
            const int row = e >> 5, pr = (e & 31) * 2;
            const int brow = (row < P_POP) ? row : (P_POP - 1);
            const float* src = B + (size_t)brow * D_DIM + kb + pr;
            __builtin_prefetch(src + KSLAB, 0, 1);               // global_prefetch next slab
            const float2 f = *(const float2*)src;
            v2h h; h.x = (_Float16)f.x; h.y = (_Float16)f.y;
            *(v2h*)(Bf + row * KSLAB + pr) = h;
        }
        __syncthreads();

        // consume slab: 3 A-frags + 2 B-frags -> 6 WMMAs per 32-K step (all unrolled)
        const _Float16* abase = Af + mloc * KSLAB;
        const _Float16* bbase = Bf + (wave * 16 + mloc) * KSLAB;
#pragma unroll
        for (int ks = 0; ks < KSLAB; ks += 32) {
            v16h afr[3], bfr[2];
#pragma unroll
            for (int tm = 0; tm < 3; ++tm) {
                const _Float16* ab = abase + (tm * 16) * KSLAB;
                *((v8h*)&afr[tm])     = *(const v8h*)(ab + ks + khiA);
                *((v8h*)&afr[tm] + 1) = *(const v8h*)(ab + ks + khiA + 8);
            }
#pragma unroll
            for (int j = 0; j < 2; ++j) {
                const _Float16* bb = bbase + (j * 8 * 16) * KSLAB;
                *((v8h*)&bfr[j])     = *(const v8h*)(bb + ks + khiB);
                *((v8h*)&bfr[j] + 1) = *(const v8h*)(bb + ks + khiB + 8);
            }
#pragma unroll
            for (int tm = 0; tm < 3; ++tm)
#pragma unroll
                for (int j = 0; j < 2; ++j)
                    acc[tm][j] = __builtin_amdgcn_wmma_f32_16x16x32_f16(
                        false, afr[tm], false, bfr[j], (short)0, acc[tm][j], false, false);
        }
        __syncthreads();   // slab fully consumed before re-staging
    }

    // ---- spill accumulators to the f32 score panel (aliases staging LDS) ----
#pragma unroll
    for (int tm = 0; tm < 3; ++tm)
#pragma unroll
        for (int j = 0; j < 2; ++j) {
            const int mbase = tm * 16 + (lane >> 4) * 8;  // C/D layout: VGPR r -> row+r
            const int ncol  = (wave + 8 * j) * 16 + (lane & 15);
#pragma unroll
            for (int r = 0; r < 8; ++r)
                S[(mbase + r) * NPAD + ncol] = acc[tm][j][r];
        }
    __syncthreads();

    // ---- per selected row: softmax normalizer + argmax + 9-neighbor fusion ----
    const int offs[9] = {0, -1, 1, -10, 10, -11, -9, 9, 11};
    for (int r = wave; r < KATTN; r += 8) {
        const float* row = &S[r * NPAD];

        float mx = -3.4e38f; int am = 0;
        for (int c = lane; c < P_POP; c += 32) {
            float v = row[c];
            if (v > mx) { mx = v; am = c; }
        }
        for (int off = 16; off; off >>= 1) {   // wave32 reduce, first-max tie rule
            float omx = __shfl_xor(mx, off);
            int   oam = __shfl_xor(am, off);
            if (omx > mx || (omx == mx && oam < am)) { mx = omx; am = oam; }
        }
        float Z = 0.f;
        for (int c = lane; c < P_POP; c += 32) Z += __expf(row[c] - mx);
        for (int off = 16; off; off >>= 1) Z += __shfl_xor(Z, off);
        const float invZ = 1.f / Z;

        const int idx = am, col = idx % 10;
        const bool up = idx > 9, dn = idx < 200;
        const bool val[9] = {true, col != 0, col != 9, up, dn,
                             up && (col > 0), up && (col < 9),
                             dn && (col > 0), dn && (col < 9)};
        int nb[9]; float w9[9];
        float m9 = -3.4e38f;
#pragma unroll
        for (int t = 0; t < 9; ++t) {
            nb[t] = idx + offs[t];
            w9[t] = val[t] ? __expf(row[nb[t]] - mx) * invZ : 0.f;
            if (val[t] && w9[t] > m9) m9 = w9[t];
        }
        float sw[9]; float z9 = 0.f;
#pragma unroll
        for (int t = 0; t < 9; ++t) {
            sw[t] = val[t] ? __expf(w9[t] - m9) : 0.f;
            z9 += sw[t];
        }
        const float inv9 = 1.f / z9;

        float* dst = O + (size_t)ridx[r] * D_DIM;
        for (int c = lane; c < D_DIM; c += 32) {
            float facc = 0.f;
#pragma unroll
            for (int t = 0; t < 9; ++t)
                if (val[t]) facc += sw[t] * B[(size_t)nb[t] * D_DIM + c];
            dst[c] = facc * inv9;
        }
    }
}

// ---------------- mutation kernel ----------------
// One workgroup per output matrix: overwrite 20 random rows with mean + std*N(0,1),
// stats over the 6 ORIGINAL block inputs (a[0..2], b[0..2]), var with ddof=1.
__global__ __launch_bounds__(256) void mut_kernel(const float* __restrict__ in,
                                                  float* __restrict__ out) {
    __shared__ int   mrow[KMUT];
    __shared__ short pool[P_POP];

    const int o    = blockIdx.x;              // 0..127 output matrix
    const int side = (o >= 64) ? 1 : 0;
    const int rem  = side ? (o - 64) : o;
    const int blk  = rem >> 2, j = rem & 3;
    const int key  = 8 + 2 * j + side;        // mutation keys k[8..15]

    const int tid = threadIdx.x;
    if (tid == 0) {
        for (int i = 0; i < P_POP; ++i) pool[i] = (short)i;
        unsigned long long s =
            splitmix64(0x5EEDF00DULL ^ ((unsigned long long)blk << 16) ^ (unsigned long long)key);
        for (int t = 0; t < KMUT; ++t) {
            s = splitmix64(s);
            int jj = t + (int)(s % (unsigned long long)(P_POP - t));
            short tmp = pool[t]; pool[t] = pool[jj]; pool[jj] = tmp;
            mrow[t] = pool[t];
        }
    }
    __syncthreads();

    const float* s0 = in + (size_t)(4 * blk + 0) * MAT_ELEMS;
    const float* s1 = in + (size_t)(4 * blk + 1) * MAT_ELEMS;
    const float* s2 = in + (size_t)(4 * blk + 2) * MAT_ELEMS;
    const float* s3 = in + (size_t)(64 + 4 * blk + 0) * MAT_ELEMS;
    const float* s4 = in + (size_t)(64 + 4 * blk + 1) * MAT_ELEMS;
    const float* s5 = in + (size_t)(64 + 4 * blk + 2) * MAT_ELEMS;
    float* dst = out + (size_t)o * MAT_ELEMS;

    const unsigned long long eseed =
        splitmix64(0xC0FFEEULL ^ ((unsigned long long)blk << 20) ^ ((unsigned long long)key << 8));

    for (int e = tid; e < KMUT * D_DIM; e += 256) {
        const int rr  = e / D_DIM;
        const int c   = e - rr * D_DIM;
        const int off = mrow[rr] * D_DIM + c;

        const float v0 = s0[off], v1 = s1[off], v2 = s2[off];
        const float v3 = s3[off], v4 = s4[off], v5 = s5[off];
        const float mean = (v0 + v1 + v2 + v3 + v4 + v5) * (1.0f / 6.0f);
        const float d0 = v0 - mean, d1 = v1 - mean, d2 = v2 - mean;
        const float d3 = v3 - mean, d4 = v4 - mean, d5 = v5 - mean;
        const float var = (d0 * d0 + d1 * d1 + d2 * d2 + d3 * d3 + d4 * d4 + d5 * d5) * 0.2f;
        const float sd  = sqrtf(var + 1e-8f);

        unsigned long long h1 = splitmix64(eseed ^ ((unsigned long long)off * 0x9E3779B97F4A7C15ULL));
        unsigned long long h2 = splitmix64(h1);
        const float u1 = u01(h1), u2 = u01(h2);
        const float z = sqrtf(-2.0f * __logf(u1)) * __cosf(6.28318530718f * u2);

        dst[off] = mean + sd * z;
    }
}

// ---------------- launch ----------------
extern "C" void kernel_launch(void* const* d_in, const int* in_sizes, int n_in,
                              void* d_out, int out_size, void* d_ws, size_t ws_size,
                              hipStream_t stream) {
    (void)in_sizes; (void)n_in; (void)d_ws; (void)ws_size; (void)out_size;
    const float* in  = (const float*)d_in[0];
    float*       out = (float*)d_out;

    attn_kernel<<<64, 256, 0, stream>>>(in, out, 0);   // first children
    attn_kernel<<<64, 256, 0, stream>>>(in, out, 1);   // second children (read stage-0 from out)
    mut_kernel <<<128, 256, 0, stream>>>(in, out);     // mutation in place
}
